// ViT_13460427505921
// MI455X (gfx1250) — matrix operand order repaired and checked by reading, not compiled, not run on hardware
//
#include <hip/hip_runtime.h>

// ---------------------------------------------------------------------------
// ViT forward for MI455X (gfx1250, wave32, WMMA f32_16x16x32_f16).
// f16 operand storage (one-time weight/activation conversion), f32 accumulate,
// f32 residual stream. Flash-attention with block-cooperative K/V staging.
// ---------------------------------------------------------------------------

typedef __attribute__((ext_vector_type(16))) _Float16 v16h;
typedef __attribute__((ext_vector_type(8)))  _Float16 v8h;
typedef __attribute__((ext_vector_type(8)))  float    v8f;

#define VDIM  512
#define SEQ   1025
#define BATCH 4
#define ROWS  (BATCH*SEQ)   // 4100 token rows

// ---------------- small helpers -------------------------------------------

__device__ inline v16h cat16(v8h lo, v8h hi) {
  v16h r;
#pragma unroll
  for (int i = 0; i < 8; i++) { r[i] = lo[i]; r[i + 8] = hi[i]; }
  return r;
}

__device__ inline v8f zero8() {
  v8f r;
#pragma unroll
  for (int i = 0; i < 8; i++) r[i] = 0.f;
  return r;
}

__device__ inline v8h zero8h() {
  v8h r;
#pragma unroll
  for (int i = 0; i < 8; i++) r[i] = (_Float16)0.f;
  return r;
}

__device__ inline v8h cvt8(float4 a, float4 b) {
  v8h r;
  r[0] = (_Float16)a.x; r[1] = (_Float16)a.y; r[2] = (_Float16)a.z; r[3] = (_Float16)a.w;
  r[4] = (_Float16)b.x; r[5] = (_Float16)b.y; r[6] = (_Float16)b.z; r[7] = (_Float16)b.w;
  return r;
}

// wave32 butterfly via ds_swizzle (group-of-32 xor mode: offset[15]=0,
// xor_mask in [14:10], and_mask=31). Immediate must be constant -> template.
template <int MASK>
__device__ inline float swz(float x) {
  return __builtin_bit_cast(float,
      __builtin_amdgcn_ds_swizzle(__builtin_bit_cast(int, x), (MASK << 10) | 31));
}
__device__ inline float redsum16(float x) {  // reduce across 16-lane half
  x += swz<1>(x); x += swz<2>(x); x += swz<4>(x); x += swz<8>(x); return x;
}
__device__ inline float redmax16(float x) {
  x = fmaxf(x, swz<1>(x)); x = fmaxf(x, swz<2>(x));
  x = fmaxf(x, swz<4>(x)); x = fmaxf(x, swz<8>(x)); return x;
}
__device__ inline float redsum32(float x) { x = redsum16(x); x += swz<16>(x); return x; }

// ---------------- one-time f32 -> f16 conversion ---------------------------

__global__ __launch_bounds__(256)
void cvt_kernel(const float* __restrict__ src, _Float16* __restrict__ dst, int n)
{
  int i = (blockIdx.x * 256 + threadIdx.x) * 8;
  if (i < n) {
    float4 a = *(const float4*)(src + i);
    float4 b = *(const float4*)(src + i + 4);
    *(v8h*)(dst + i) = cvt8(a, b);
  }
}

// ---------------- generic f16-WMMA GEMM ------------------------------------
// C[M,N] = A[M,K] @ W[K,N]  (+ bias[N]) (+ resid[M,N]) (GELU)
// A, W pre-converted f16. Block: 256 threads (8 waves), tile 128x128,
// k-step 32, software-pipelined: next tile's global loads issue while the
// current tile is consumed by WMMAs. LDS pitch 40 halfs keeps all fragment
// reads 16B-aligned b128 and conflict-free.
// EPI: 0 = none, 1 = +bias, 2 = +bias+resid, 3 = +bias+GELU(exact)
// HOUT: store f16 (Ch) instead of f32 (Cf).

template <int EPI, bool HOUT>
__global__ __launch_bounds__(256)
void gemm_kernel(const _Float16* __restrict__ A, const _Float16* __restrict__ W,
                 const float* __restrict__ bias, const float* __restrict__ R,
                 float* __restrict__ Cf, _Float16* __restrict__ Ch,
                 int M, int N, int K)
{
  __shared__ __align__(16) _Float16 As[128 * 40];
  __shared__ __align__(16) _Float16 Bs[128 * 40];

  const int tid  = threadIdx.x;
  const int lane = tid & 31;
  const int wid  = tid >> 5;
  const int wm   = wid & 1;      // 0..1  -> 64-row half
  const int wn   = wid >> 1;     // 0..3  -> 32-col quarter
  const int lm   = lane & 15;
  const int lh   = lane >> 4;    // 0/1 half-wave
  const int bm   = blockIdx.y * 128;
  const int bn   = blockIdx.x * 128;

  // staging coordinates
  const int ar  = tid >> 1,  ac0 = (tid & 1) * 16;   // A: row 0..127, col 0/16
  const int bk  = tid >> 3,  bn0 = (tid & 7) * 16;   // B: k 0..31, n0 0..112
  const bool arow_ok = (bm + ar) < M;
  const _Float16* Ap = A + (size_t)(bm + ar) * K + ac0;
  const _Float16* Wp = W + (size_t)bk * N + bn + bn0;

  v8f acc[4][2];
#pragma unroll
  for (int i = 0; i < 4; i++)
#pragma unroll
    for (int j = 0; j < 2; j++) acc[i][j] = zero8();

  // prologue: stage tile k0 = 0 into registers
  v8h a0, a1, b0, b1;
  a0 = arow_ok ? *(const v8h*)(Ap)     : zero8h();
  a1 = arow_ok ? *(const v8h*)(Ap + 8) : zero8h();
  b0 = *(const v8h*)(Wp);
  b1 = *(const v8h*)(Wp + 8);

  for (int k0 = 0; k0 < K; k0 += 32) {
    // --- spill staged tile to LDS ----------------------------------------
    *(v8h*)&As[ar * 40 + ac0]     = a0;
    *(v8h*)&As[ar * 40 + ac0 + 8] = a1;
#pragma unroll
    for (int j = 0; j < 8; j++) {   // transpose store: Bs[n][k]
      Bs[(bn0 + j)     * 40 + bk] = b0[j];
      Bs[(bn0 + 8 + j) * 40 + bk] = b1[j];
    }
    __syncthreads();

    // --- issue next tile's global loads (overlap with WMMA below) --------
    if (k0 + 32 < K) {
      a0 = arow_ok ? *(const v8h*)(Ap + k0 + 32)     : zero8h();
      a1 = arow_ok ? *(const v8h*)(Ap + k0 + 32 + 8) : zero8h();
      b0 = *(const v8h*)(Wp + (size_t)(k0 + 32) * N);
      b1 = *(const v8h*)(Wp + (size_t)(k0 + 32) * N + 8);
      if (k0 + 64 < K)   // prefetch the tile after that (global_prefetch_b8)
        __builtin_prefetch(Wp + (size_t)(k0 + 64) * N, 0, 1);
    }

    // --- fragments per ISA 16-bit layouts ---------------------------------
    // A: lane row = lm; lanes 0-15 hold K 0-7/16-23, lanes 16-31 K 8-15/24-31
    v16h af[4], bf[2];
#pragma unroll
    for (int mt = 0; mt < 4; mt++) {
      int row = wm * 64 + mt * 16 + lm;
      af[mt] = cat16(*(v8h*)&As[row * 40 + lh * 8],
                     *(v8h*)&As[row * 40 + lh * 8 + 16]);
    }
    // B: lane col = lm; lanes 0-15 hold K 0-7/16-23, lanes 16-31 K 8-15/24-31
#pragma unroll
    for (int nt = 0; nt < 2; nt++) {
      int col = wn * 32 + nt * 16 + lm;
      bf[nt] = cat16(*(v8h*)&Bs[col * 40 + lh * 8],
                     *(v8h*)&Bs[col * 40 + lh * 8 + 16]);
    }
#pragma unroll
    for (int mt = 0; mt < 4; mt++)
#pragma unroll
      for (int nt = 0; nt < 2; nt++)
        acc[mt][nt] = __builtin_amdgcn_wmma_f32_16x16x32_f16(
            false, af[mt], false, bf[nt], (short)0, acc[mt][nt], false, false);
    __syncthreads();
  }

  // --- epilogue: C/D layout row = lh*8 + v, col = lm ------------------------
#pragma unroll
  for (int mt = 0; mt < 4; mt++)
#pragma unroll
    for (int nt = 0; nt < 2; nt++) {
      int col = bn + wn * 32 + nt * 16 + lm;
      float bv = (EPI >= 1) ? bias[col] : 0.f;
#pragma unroll
      for (int v = 0; v < 8; v++) {
        int row = bm + wm * 64 + mt * 16 + lh * 8 + v;
        if (row < M) {
          float x = acc[mt][nt][v] + bv;
          if (EPI == 2) x += R[(size_t)row * N + col];
          if (EPI == 3) x = 0.5f * x * (1.f + erff(x * 0.70710678118f));
          if (HOUT) Ch[(size_t)row * N + col] = (_Float16)x;
          else      Cf[(size_t)row * N + col] = x;
        }
      }
    }
}

// ---------------- token assembly: X[b][s] = (cls | emb) + pos ---------------

__global__ __launch_bounds__(256)
void assemble_kernel(const float* __restrict__ emb, const float* __restrict__ cls,
                     const float* __restrict__ pos, float* __restrict__ X)
{
  int idx = blockIdx.x * 256 + threadIdx.x;   // over ROWS*VDIM exactly
  int d = idx & (VDIM - 1);
  int t = idx >> 9;
  int b = t / SEQ;
  int s = t - b * SEQ;
  float v = (s == 0) ? cls[d] : emb[((size_t)b * 1024 + (s - 1)) * VDIM + d];
  X[idx] = v + pos[s * VDIM + d];
}

// ---------------- layernorm (512) -> f16, one wave per row -----------------

__global__ __launch_bounds__(256)
void layernorm_kernel(const float* __restrict__ X, const float* __restrict__ s,
                      const float* __restrict__ b, _Float16* __restrict__ Y, int rows)
{
  int lane = threadIdx.x & 31, wid = threadIdx.x >> 5;
  int row = blockIdx.x * 8 + wid;
  if (row >= rows) return;
  const int c0 = lane * 16;
  const float* xr = X + (size_t)row * VDIM + c0;
  float vals[16], sum = 0.f, sq = 0.f;
#pragma unroll
  for (int q = 0; q < 4; q++) {
    float4 f = *(const float4*)(xr + q * 4);
    vals[q*4+0] = f.x; vals[q*4+1] = f.y; vals[q*4+2] = f.z; vals[q*4+3] = f.w;
  }
#pragma unroll
  for (int i = 0; i < 16; i++) { sum += vals[i]; sq += vals[i] * vals[i]; }
  sum = redsum32(sum); sq = redsum32(sq);
  float mean = sum * (1.f / 512.f);
  float rs = rsqrtf(sq * (1.f / 512.f) - mean * mean + 1e-5f);
  v8h o0, o1;
#pragma unroll
  for (int i = 0; i < 8; i++) {
    o0[i] = (_Float16)((vals[i]     - mean) * rs * s[c0 + i]     + b[c0 + i]);
    o1[i] = (_Float16)((vals[i + 8] - mean) * rs * s[c0 + i + 8] + b[c0 + i + 8]);
  }
  _Float16* yr = Y + (size_t)row * VDIM + c0;
  *(v8h*)(yr)     = o0;
  *(v8h*)(yr + 8) = o1;
}

// ---------------- flash attention (f16 QKV, cooperative K/V staging) -------
// QKV16: [ROWS][1536] halfs (Q | K | V, each 8 heads x 64). All 4 waves of a
// block work on the SAME (b,h) with 4 adjacent 16-row q-tiles, so each 32-key
// K/V chunk is loaded from global ONCE per block into LDS (4x traffic cut)
// and consumed by all waves. Grid = 32 (b,h) x 17 tile-groups = 544 blocks.
// Waves whose q-tile exceeds 64 recompute tile 64 (EXEC stays all-ones for
// WMMA) but skip their stores. 1/sqrt(d) applied to scores.

__global__ __launch_bounds__(128)
void attn_kernel(const _Float16* __restrict__ QKV, _Float16* __restrict__ O)
{
  __shared__ __align__(16) _Float16 Kl[32 * 72];      // K:  32 keys x 64 d
  __shared__ __align__(16) _Float16 Vt[64 * 40];      // V^T: 64 d x 32 keys
  __shared__ __align__(16) _Float16 Pl[4][16 * 40];   // per-wave P: 16 x 32

  const int tid  = threadIdx.x;
  const int lane = tid & 31, wid = tid >> 5;
  const int lm = lane & 15, lh = lane >> 4;
  const int bh  = blockIdx.x / 17;        // 0..31
  const int grp = blockIdx.x % 17;        // 0..16
  const int b = bh >> 3, h = bh & 7;
  const int qt = grp * 4 + wid;           // 0..67
  const bool live = (qt < 65);
  const int qtc = live ? qt : 64;

  const size_t rowbase = (size_t)b * SEQ;

  // --- Q fragments: direct aligned 16B loads --------------------------------
  v16h qf[2];
  {
    int s = qtc * 16 + lm; if (s > 1024) s = 1024;     // clamp pad rows
    const _Float16* qp = QKV + (rowbase + s) * 1536 + h * 64;
#pragma unroll
    for (int kc = 0; kc < 2; kc++)
      qf[kc] = cat16(*(const v8h*)(qp + kc * 32 + lh * 8),
                     *(const v8h*)(qp + kc * 32 + lh * 8 + 16));
  }

  float m[8], lsum[8];
  v8f oacc[4];
#pragma unroll
  for (int v = 0; v < 8; v++) { m[v] = -1e30f; lsum[v] = 0.f; }
#pragma unroll
  for (int t = 0; t < 4; t++) oacc[t] = zero8();

  _Float16* P = &Pl[wid][0];

  // cooperative staging coords: thread -> (key 0..31, 16-d quarter)
  const int skey = tid >> 2, sdq = (tid & 3) * 16;

  for (int kb = 0; kb < SEQ; kb += 32) {       // 33 chunks cover keys 0..1055
    __syncthreads();   // previous chunk fully consumed before overwrite
    // --- stage K chunk (row-major) and V chunk (transposed) once ----------
    {
      int gk = kb + skey;
      int gkc = gk > 1024 ? 1024 : gk;
      const _Float16* kp = QKV + (rowbase + gkc) * 1536 + 512 + h * 64 + sdq;
      v8h k0 = *(const v8h*)(kp), k1 = *(const v8h*)(kp + 8);
      *(v8h*)&Kl[skey * 72 + sdq]     = k0;
      *(v8h*)&Kl[skey * 72 + sdq + 8] = k1;
      const _Float16* vp = QKV + (rowbase + gkc) * 1536 + 1024 + h * 64 + sdq;
      v8h v0 = *(const v8h*)(vp), v1 = *(const v8h*)(vp + 8);
#pragma unroll
      for (int j = 0; j < 8; j++) {
        Vt[(sdq + j)     * 40 + skey] = v0[j];
        Vt[(sdq + 8 + j) * 40 + skey] = v1[j];
      }
    }
    __syncthreads();

    // --- scores S = (Q K^T) * scale : two 16-key sub-tiles, K from LDS ----
    v8f sacc[2];
#pragma unroll
    for (int nt = 0; nt < 2; nt++) {
      int lk = nt * 16 + lm;                    // local key 0..31
      v16h kf0 = cat16(*(v8h*)&Kl[lk * 72 + lh * 8],
                       *(v8h*)&Kl[lk * 72 + lh * 8 + 16]);
      v16h kf1 = cat16(*(v8h*)&Kl[lk * 72 + 32 + lh * 8],
                       *(v8h*)&Kl[lk * 72 + 32 + lh * 8 + 16]);
      v8f a = zero8();
      a = __builtin_amdgcn_wmma_f32_16x16x32_f16(false, qf[0], false, kf0,
                                                 (short)0, a, false, false);
      a = __builtin_amdgcn_wmma_f32_16x16x32_f16(false, qf[1], false, kf1,
                                                 (short)0, a, false, false);
      int key = kb + lk;
#pragma unroll
      for (int v = 0; v < 8; v++)
        a[v] = (key > 1024) ? -1e30f : a[v] * 0.125f;   // scale + pad mask
      sacc[nt] = a;
    }
    // --- online softmax per row (row = lh*8+v; columns across 16 lanes) ---
#pragma unroll
    for (int v = 0; v < 8; v++) {
      float mx = redmax16(fmaxf(sacc[0][v], sacc[1][v]));
      float mnew = fmaxf(m[v], mx);
      float alpha = __expf(m[v] - mnew);
      float e0 = __expf(sacc[0][v] - mnew);
      float e1 = __expf(sacc[1][v] - mnew);
      lsum[v] = lsum[v] * alpha + redsum16(e0 + e1);
      m[v] = mnew;
#pragma unroll
      for (int t = 0; t < 4; t++) oacc[t][v] *= alpha;
      sacc[0][v] = e0; sacc[1][v] = e1;
    }
    // --- P (C-layout) -> per-wave LDS -> re-read as A fragment ------------
#pragma unroll
    for (int nt = 0; nt < 2; nt++)
#pragma unroll
      for (int v = 0; v < 8; v++)
        P[(lh * 8 + v) * 40 + nt * 16 + lm] = (_Float16)sacc[nt][v];

    v16h pf = cat16(*(v8h*)&P[lm * 40 + lh * 8],
                    *(v8h*)&P[lm * 40 + lh * 8 + 16]);
#pragma unroll
    for (int t = 0; t < 4; t++) {
      int d = t * 16 + lm;
      v16h vf = cat16(*(v8h*)&Vt[d * 40 + lh * 8],
                      *(v8h*)&Vt[d * 40 + lh * 8 + 16]);
      oacc[t] = __builtin_amdgcn_wmma_f32_16x16x32_f16(
                    false, pf, false, vf, (short)0, oacc[t], false, false);
    }
  }

  // --- normalize + write O (f16); duplicate waves skip --------------------
  if (live) {
#pragma unroll
    for (int t = 0; t < 4; t++)
#pragma unroll
      for (int v = 0; v < 8; v++) {
        int s = qt * 16 + lh * 8 + v;
        if (s < SEQ)
          O[(rowbase + s) * VDIM + h * 64 + t * 16 + lm] =
              (_Float16)(oacc[t][v] / lsum[v]);
      }
  }
}

// ---------------- classifier head: LN -> 512x2 -> sigmoid ------------------

__global__ __launch_bounds__(128)
void head_kernel(const float* __restrict__ X, const float* __restrict__ s,
                 const float* __restrict__ b, const float* __restrict__ w,
                 const float* __restrict__ hb, float* __restrict__ out)
{
  int lane = threadIdx.x & 31, wid = threadIdx.x >> 5;   // wid = batch
  const float* xr = X + (size_t)wid * SEQ * VDIM;        // token 0 of batch
  float vals[16], sum = 0.f, sq = 0.f;
#pragma unroll
  for (int i = 0; i < 16; i++) {
    float v = xr[lane + i * 32]; vals[i] = v; sum += v; sq += v * v;
  }
  sum = redsum32(sum); sq = redsum32(sq);
  float mean = sum * (1.f / 512.f);
  float rs = rsqrtf(sq * (1.f / 512.f) - mean * mean + 1e-5f);
  float xn[16];
#pragma unroll
  for (int i = 0; i < 16; i++) {
    int c = lane + i * 32;
    xn[i] = (vals[i] - mean) * rs * s[c] + b[c];
  }
#pragma unroll
  for (int c = 0; c < 2; c++) {
    float dot = 0.f;
#pragma unroll
    for (int i = 0; i < 16; i++) dot += xn[i] * w[(lane + i * 32) * 2 + c];
    dot = redsum32(dot);
    if (lane == 0) out[wid * 2 + c] = 1.f / (1.f + __expf(-(dot + hb[c])));
  }
}

// ---------------- host-side orchestration ----------------------------------

extern "C" void kernel_launch(void* const* d_in, const int* in_sizes, int n_in,
                              void* d_out, int out_size, void* d_ws, size_t ws_size,
                              hipStream_t stream)
{
  const float* img     = (const float*)d_in[0];
  const float* embed_w = (const float*)d_in[1];
  const float* embed_b = (const float*)d_in[2];
  const float* pos_emb = (const float*)d_in[3];
  const float* cls_tok = (const float*)d_in[4];
  const float* ln1_s   = (const float*)d_in[5];
  const float* ln1_b   = (const float*)d_in[6];
  const float* qkv_w   = (const float*)d_in[7];
  const float* out_w   = (const float*)d_in[8];
  const float* out_b   = (const float*)d_in[9];
  const float* ln2_s   = (const float*)d_in[10];
  const float* ln2_b   = (const float*)d_in[11];
  const float* ff_w1   = (const float*)d_in[12];
  const float* ff_b1   = (const float*)d_in[13];
  const float* ff_w2   = (const float*)d_in[14];
  const float* ff_b2   = (const float*)d_in[15];
  const float* hl_s    = (const float*)d_in[16];
  const float* hl_b    = (const float*)d_in[17];
  const float* head_w  = (const float*)d_in[18];
  const float* head_b  = (const float*)d_in[19];

  // Workspace layout: f32 residual X, f16 everything else. The embed GEMM's
  // f32 output aliases Sc (live ranges disjoint). Total ~89 MB.
  char* p = (char*)d_ws;
  float*    X     = (float*)p;    p += (size_t)ROWS * VDIM * 4;
  _Float16* Hb    = (_Float16*)p; p += (size_t)ROWS * VDIM * 2;
  _Float16* Sc    = (_Float16*)p; p += (size_t)ROWS * 2048 * 2;  // QKV / FF hidden
  _Float16* Ob    = (_Float16*)p; p += (size_t)ROWS * VDIM * 2;
  _Float16* img16 = (_Float16*)p; p += (size_t)4096 * VDIM * 2;
  _Float16* w_emb = (_Float16*)p; p += (size_t)512 * 512 * 2;
  _Float16* w_qkv = (_Float16*)p; p += (size_t)8 * 512 * 1536 * 2;
  _Float16* w_out = (_Float16*)p; p += (size_t)8 * 512 * 512 * 2;
  _Float16* w_f1  = (_Float16*)p; p += (size_t)8 * 512 * 2048 * 2;
  _Float16* w_f2  = (_Float16*)p; p += (size_t)8 * 2048 * 512 * 2;
  float* embT = (float*)Sc;   // alias: embed GEMM f32 output (8.4MB < 16.8MB)

  // 0) one-time f32 -> f16 conversions (all counts multiples of 2048)
  cvt_kernel<<<2097152 / 2048, 256, 0, stream>>>(img,     img16, 2097152);
  cvt_kernel<<< 262144 / 2048, 256, 0, stream>>>(embed_w, w_emb,  262144);
  cvt_kernel<<<6291456 / 2048, 256, 0, stream>>>(qkv_w,   w_qkv, 6291456);
  cvt_kernel<<<2097152 / 2048, 256, 0, stream>>>(out_w,   w_out, 2097152);
  cvt_kernel<<<8388608 / 2048, 256, 0, stream>>>(ff_w1,   w_f1,  8388608);
  cvt_kernel<<<8388608 / 2048, 256, 0, stream>>>(ff_w2,   w_f2,  8388608);

  // 1) patch embedding: embT = img @ embed_w + embed_b   (f32 out)
  gemm_kernel<1, false><<<dim3(VDIM / 128, 4096 / 128), 256, 0, stream>>>(
      img16, w_emb, embed_b, nullptr, embT, nullptr, 4096, VDIM, 512);
  // 2) tokens: X = concat(cls, emb) + pos
  assemble_kernel<<<(ROWS * VDIM) / 256, 256, 0, stream>>>(embT, cls_tok, pos_emb, X);

  for (int l = 0; l < 8; l++) {
    layernorm_kernel<<<(ROWS + 7) / 8, 256, 0, stream>>>(
        X, ln1_s + l * VDIM, ln1_b + l * VDIM, Hb, ROWS);
    gemm_kernel<0, true><<<dim3(1536 / 128, (ROWS + 127) / 128), 256, 0, stream>>>(
        Hb, w_qkv + (size_t)l * VDIM * 1536, nullptr, nullptr, nullptr, Sc,
        ROWS, 1536, VDIM);
    attn_kernel<<<32 * 17, 128, 0, stream>>>(Sc, Ob);
    gemm_kernel<2, false><<<dim3(VDIM / 128, (ROWS + 127) / 128), 256, 0, stream>>>(
        Ob, w_out + (size_t)l * VDIM * VDIM, out_b + l * VDIM, X, X, nullptr,
        ROWS, VDIM, VDIM);
    layernorm_kernel<<<(ROWS + 7) / 8, 256, 0, stream>>>(
        X, ln2_s + l * VDIM, ln2_b + l * VDIM, Hb, ROWS);
    gemm_kernel<3, true><<<dim3(2048 / 128, (ROWS + 127) / 128), 256, 0, stream>>>(
        Hb, w_f1 + (size_t)l * VDIM * 2048, ff_b1 + l * 2048, nullptr, nullptr, Sc,
        ROWS, 2048, VDIM);
    gemm_kernel<2, false><<<dim3(VDIM / 128, (ROWS + 127) / 128), 256, 0, stream>>>(
        Sc, w_f2 + (size_t)l * 2048 * VDIM, ff_b2 + l * VDIM, X, X, nullptr,
        ROWS, VDIM, 2048);
  }

  head_kernel<<<1, 128, 0, stream>>>(X, hl_s, hl_b, head_w, head_b, (float*)d_out);
}
